// FraudDetectionHybrid_65481071400297
// MI455X (gfx1250) — matrix operand order repaired and checked by reading, not compiled, not run on hardware
//
#include <hip/hip_runtime.h>

// ---- CDNA5 (gfx1250) wave32 WMMA types ----
typedef __attribute__((ext_vector_type(16))) __bf16 v16bf;
typedef __attribute__((ext_vector_type(8)))  float  v8f;

#define B_BATCH 4096
#define S_SEQ   512
#define H_DIM   32
#define D_DIM   2
#define L_EXT   3
#define KDIM    (D_DIM + H_DIM)        // 34
#define WAVES_PER_WG 4
#define ROWS_PER_WAVE 16
#define ROWS_PER_WG (WAVES_PER_WG * ROWS_PER_WAVE)   // 64
#define BLOCK_T (WAVES_PER_WG * 32)                  // 128

struct alignas(16) U4 { unsigned a, b, c, d; };
union Frag  { U4 q[2]; v16bf v; };
union Pack2 { __bf16 h[2]; unsigned u; };

// Fast nonlinearities: v_exp_f32 + v_rcp_f32 (TRANS ops, co-execute with XDL WMMA).
__device__ __forceinline__ float fast_sigmoid(float x) {
  return __builtin_amdgcn_rcpf(1.0f + __expf(-x));
}
__device__ __forceinline__ float fast_tanh(float x) {
  // tanh(x) = 1 - 2/(e^{2x}+1); saturates correctly at +/-1.
  return 1.0f - 2.0f * __builtin_amdgcn_rcpf(__expf(2.0f * x) + 1.0f);
}
__device__ __forceinline__ unsigned pack_pair(const float* gW, int n, int k) {
  float f0 = (k     < KDIM) ? gW[n * KDIM + k    ] : 0.0f;
  float f1 = (k + 1 < KDIM) ? gW[n * KDIM + k + 1] : 0.0f;
  Pack2 p; p.h[0] = (__bf16)f0; p.h[1] = (__bf16)f1;
  return p.u;
}
__device__ __forceinline__ v8f splat8(float s) {
  v8f r = {s, s, s, s, s, s, s, s};
  return r;
}
// Zero-instruction ordering point: stops the compiler from moving LDS ops across
// it; hardware per-wave in-order DS execution provides the actual visibility.
__device__ __forceinline__ void wave_fence() {
  __asm__ volatile("" ::: "memory");
  __builtin_amdgcn_wave_barrier();
}

// Fused photonic-extractor + LSTM + classifier.
// Each wave owns 16 batch rows; hx/cx stay on-chip for all 512 steps.
// z = [xt(2) | hx(32)] @ Wcat^T + b, K padded to 64 -> 2x v_wmma_f32_16x16x32_bf16
// per N-tile, 8 N-tiles -> 16 WMMAs per step per wave. All per-step LDS state is
// wave-private, so the recurrence needs no workgroup barriers.
__global__ __launch_bounds__(BLOCK_T)
void fraud_lstm_wmma(const float* __restrict__ x,     // [B,S,D]
                     const float* __restrict__ extW,  // [3,2,2]
                     const float* __restrict__ extb,  // [3,2]
                     const float* __restrict__ extsc, // [3,2]
                     const float* __restrict__ extsh, // [3,2]
                     const float* __restrict__ gW,    // [128,34]  (Wcat)
                     const float* __restrict__ gb,    // [128]
                     const float* __restrict__ Wc,    // [1,32]
                     const float* __restrict__ bc,    // [1]
                     float* __restrict__ out)         // [B,1]
{
  // B fragments pre-swizzled to exact 16-bit B 32x16 lane layout:
  // frag f = kc*8+nt ; half hf: lanes 0-15 hold K=16*hf+0..7 of chunk (pairs), lanes 16-31 K=16*hf+8..15.
  __shared__ U4 sB[16][2][32];                                // 16 KB, read-only after init
  __shared__ alignas(16) __bf16 aRows[WAVES_PER_WG][16][64];  // 8 KB wave-private A staging
  __shared__ float sWc[H_DIM];

  const int tid  = threadIdx.x;
  const int w    = tid >> 5;
  const int lane = tid & 31;
  const int hsel = lane >> 4;     // which K-half of a chunk this lane holds
  const int l16  = lane & 15;
  const int rowBase = blockIdx.x * ROWS_PER_WG + w * ROWS_PER_WAVE;

  // ---- stage Wcat^T (bf16) into fragment-layout LDS ----
  for (int e = tid; e < 16 * 2 * 32; e += BLOCK_T) {
    int ln = e & 31;
    int hf = (e >> 5) & 1;
    int f  = e >> 6;                 // 0..15
    int kc = f >> 3, nt = f & 7;
    int n  = nt * 16 + (ln & 15);
    int kb = kc * 32 + 8 * (ln >> 4) + 16 * hf;   // first K of this lane's 4 dword pairs
    U4 r;
    r.a = pack_pair(gW, n, kb + 0);
    r.b = pack_pair(gW, n, kb + 2);
    r.c = pack_pair(gW, n, kb + 4);
    r.d = pack_pair(gW, n, kb + 6);
    sB[f][hf][ln] = r;
  }
  // zero A staging (h0 = 0, K>=34 pad stays 0 forever)
  unsigned* az = (unsigned*)aRows;
  for (int e = tid; e < (int)(sizeof(aRows) / 4); e += BLOCK_T) az[e] = 0u;
  if (tid < H_DIM) sWc[tid] = Wc[tid];
  __syncthreads();   // one real barrier: sB shared across all 4 waves

  // per-lane bias: z column n = nt*16 + l16 (constant across the M rows a lane holds)
  float bt[8];
#pragma unroll
  for (int nt = 0; nt < 8; ++nt) bt[nt] = gb[nt * 16 + l16];

  // extractor params (uniform)
  float eW[L_EXT][2][2], eb_[L_EXT][2], esc[L_EXT][2], esh[L_EXT][2];
#pragma unroll
  for (int l = 0; l < L_EXT; ++l) {
#pragma unroll
    for (int r = 0; r < 2; ++r) {
      eW[l][r][0] = extW[l * 4 + r * 2 + 0];
      eW[l][r][1] = extW[l * 4 + r * 2 + 1];
      eb_[l][r]   = extb[l * 2 + r];
      esc[l][r]   = extsc[l * 2 + r];
      esh[l][r]   = extsh[l * 2 + r];
    }
  }

  v8f cstate[2] = { splat8(0.0f), splat8(0.0f) };   // cx, C-layout tiles (N 0..15 / 16..31)
  __bf16* myRow = &aRows[w][l16][0];

  for (int t = 0; t < S_SEQ; ++t) {
    // --- fused extractor: produce xt features for this wave's 16 rows ---
    if (lane < 16) {
      const float* xp = x + ((size_t)(rowBase + l16) * S_SEQ + t) * D_DIM;
      float f0 = xp[0], f1 = xp[1];
      __builtin_prefetch(xp + 16 * D_DIM, 0, 1);   // global_prefetch_b8, ~16 steps ahead
#pragma unroll
      for (int l = 0; l < L_EXT; ++l) {
        float n0 = fast_tanh(f0 * eW[l][0][0] + f1 * eW[l][0][1] + eb_[l][0]) * esc[l][0] + esh[l][0];
        float n1 = fast_tanh(f0 * eW[l][1][0] + f1 * eW[l][1][1] + eb_[l][1]) * esc[l][1] + esh[l][1];
        f0 = n0; f1 = n1;
      }
      myRow[0] = (__bf16)f0;
      myRow[1] = (__bf16)f1;
    }
    // x(t) + h(t-1) stores precede the fragment loads in wave program order;
    // per-wave in-order DS makes them visible across lanes without a barrier.
    wave_fence();

    // --- load A fragments per 16-bit A 16x32 lane layout ---
    const __bf16* arow = &aRows[w][l16][0];
    Frag a0, a1;
    a0.q[0] = *(const U4*)(arow + hsel * 8);            // chunk0 K 0-7 / 8-15
    a0.q[1] = *(const U4*)(arow + hsel * 8 + 16);       // chunk0 K 16-23 / 24-31
    a1.q[0] = *(const U4*)(arow + 32 + hsel * 8);       // chunk1 K 32-39 / 40-47
    a1.q[1] = *(const U4*)(arow + 32 + hsel * 8 + 16);  // chunk1 K 48-55 / 56-63

    // --- z = A @ B + bias : 16 WMMAs (8 N-tiles x 2 K-chunks) ---
    v8f acc[8];
#pragma unroll
    for (int nt = 0; nt < 8; ++nt) {
      Frag b0, b1;
      b0.q[0] = sB[nt][0][lane];
      b0.q[1] = sB[nt][1][lane];
      b1.q[0] = sB[8 + nt][0][lane];
      b1.q[1] = sB[8 + nt][1][lane];
      v8f a = splat8(bt[nt]);
      a = __builtin_amdgcn_wmma_f32_16x16x32_bf16(false, a0.v, false, b0.v, (short)0, a, false, false);
      a = __builtin_amdgcn_wmma_f32_16x16x32_bf16(false, a1.v, false, b1.v, (short)0, a, false, false);
      acc[nt] = a;
    }

    // --- gates (lane-local VALU/TRANS; co-executes with XDL) ---
    // tiles: f = acc[0..1], i = acc[2..3], g = acc[4..5], o = acc[6..7]
    v8f hnew[2];
#pragma unroll
    for (int p = 0; p < 2; ++p) {
#pragma unroll
      for (int e = 0; e < 8; ++e) {
        float fg = fast_sigmoid(acc[0 + p][e]);
        float ig = fast_sigmoid(acc[2 + p][e]);
        float gg = fast_tanh(acc[4 + p][e]);
        float og = fast_sigmoid(acc[6 + p][e]);
        float cn = fg * cstate[p][e] + ig * gg;
        cstate[p][e] = cn;
        hnew[p][e] = og * fast_tanh(cn);
      }
    }

    // --- scatter h (C-layout -> row-major bf16 A staging for next step) ---
#pragma unroll
    for (int p = 0; p < 2; ++p) {
#pragma unroll
      for (int e = 0; e < 8; ++e) {
        int m = e + 8 * hsel;
        int n = p * 16 + l16;
        aRows[w][m][2 + n] = (__bf16)hnew[p][e];
      }
    }
    wave_fence();
  }

  // --- classifier on final h (wave-private data; no block barrier needed) ---
  if (lane < 16) {
    float s = bc[0];
#pragma unroll
    for (int n = 0; n < H_DIM; ++n)
      s += sWc[n] * (float)aRows[w][l16][2 + n];
    out[rowBase + l16] = s;
  }
}

extern "C" void kernel_launch(void* const* d_in, const int* in_sizes, int n_in,
                              void* d_out, int out_size, void* d_ws, size_t ws_size,
                              hipStream_t stream) {
  (void)in_sizes; (void)n_in; (void)out_size; (void)d_ws; (void)ws_size;
  const float* x     = (const float*)d_in[0];
  const float* extW  = (const float*)d_in[1];
  const float* extb  = (const float*)d_in[2];
  const float* extsc = (const float*)d_in[3];
  const float* extsh = (const float*)d_in[4];
  const float* gW    = (const float*)d_in[5];
  const float* gb    = (const float*)d_in[6];
  const float* Wc    = (const float*)d_in[7];
  const float* bc    = (const float*)d_in[8];
  float* out = (float*)d_out;

  dim3 grid(B_BATCH / ROWS_PER_WG);   // 64 workgroups
  dim3 block(BLOCK_T);                // 128 threads = 4 wave32
  fraud_lstm_wmma<<<grid, block, 0, stream>>>(x, extW, extb, extsc, extsh,
                                              gW, gb, Wc, bc, out);
}
